// LSTM_13159779795583
// MI455X (gfx1250) — matrix operand order repaired and checked by reading, not compiled, not run on hardware
//
#include <hip/hip_runtime.h>
#include <math.h>

// ---------------------------------------------------------------------------
// 2-layer LSTM (B=32,T=512,IN=256,H=1024) + FC(1024->256) for MI455X gfx1250.
//  - parallel GEMMs: 32x64-per-wave WMMA f16, A panel staged in LDS per block
//  - recurrence: persistent 32-block kernel; W_hh in 256 VGPRs/wave (no B
//    loads on the serial path); h(t-1) staged in LDS once per block per step
//    (bank-conflict-free padded layout); cell state in registers; one grid
//    barrier per timestep. Activations time-major [t][b][.].
// ---------------------------------------------------------------------------

typedef __attribute__((ext_vector_type(16))) _Float16 v16h;
typedef __attribute__((ext_vector_type(8)))  _Float16 v8h;
typedef __attribute__((ext_vector_type(8)))  float    v8f;

#define BATCH  32
#define SEQT   512
#define INDIM  256
#define HDIM   1024
#define OUTDIM 256
#define GATES  4096   // 4*HDIM
#define HPAD   1032   // HDIM + 8 halves row padding (bank-conflict-free)
#define KCH    256    // GEMM K staging chunk (halves)
#define APAD   264    // KCH + 8

// Gate-column permutation so block nb owns hidden units [nb*32, nb*32+32)
// with all four gates contiguous:
// prow = blk*128 + gate*32 + jl   <->   srow = gate*HDIM + blk*32 + jl
__device__ __forceinline__ int perm_src_row(int prow) {
  int blk  = prow >> 7;
  int rem  = prow & 127;
  int gate = rem >> 5;
  int jl   = rem & 31;
  return gate * HDIM + blk * 32 + jl;
}

__device__ __forceinline__ float sigmoidf_(float x) {
  return 1.0f / (1.0f + __expf(-x));
}

// ------------------------------ pack kernels ------------------------------

__global__ void k_init_barrier(unsigned* bar) { bar[0] = 0u; bar[1] = 0u; }

__global__ void k_pack_f16(const float* __restrict__ src,
                           _Float16* __restrict__ dst, size_t n) {
  size_t stride = (size_t)gridDim.x * blockDim.x;
  for (size_t i = (size_t)blockIdx.x * blockDim.x + threadIdx.x; i < n; i += stride)
    dst[i] = (_Float16)src[i];
}

// x [b][t][k] (f32) -> x16 [t][b][k] (f16), time-major
__global__ void k_pack_x_tmajor(const float* __restrict__ src,
                                _Float16* __restrict__ dst) {
  size_t n = (size_t)BATCH * SEQT * INDIM;
  size_t stride = (size_t)gridDim.x * blockDim.x;
  for (size_t i = (size_t)blockIdx.x * blockDim.x + threadIdx.x; i < n; i += stride) {
    int k   = (int)(i % INDIM);
    int row = (int)(i / INDIM);        // dst row = t*BATCH + b
    int t   = row / BATCH;
    int b   = row % BATCH;
    dst[i] = (_Float16)src[((size_t)b * SEQT + t) * INDIM + k];
  }
}

__global__ void k_pack_w_perm(const float* __restrict__ src,
                              _Float16* __restrict__ dst, int K) {
  size_t n = (size_t)GATES * K;
  size_t stride = (size_t)gridDim.x * blockDim.x;
  for (size_t i = (size_t)blockIdx.x * blockDim.x + threadIdx.x; i < n; i += stride) {
    int prow = (int)(i / (size_t)K);
    int k    = (int)(i - (size_t)prow * K);
    dst[i] = (_Float16)src[(size_t)perm_src_row(prow) * K + k];
  }
}

__global__ void k_pack_bias(const float* __restrict__ bih,
                            const float* __restrict__ bhh,
                            float* __restrict__ dst) {
  int p = blockIdx.x * blockDim.x + threadIdx.x;
  if (p < GATES) {
    int s = perm_src_row(p);
    dst[p] = bih[s] + bhh[s];
  }
}

// ------------------------------ grid barrier ------------------------------

__device__ __forceinline__ void grid_barrier(unsigned* bar, unsigned nblocks) {
  __syncthreads();
  if (threadIdx.x == 0) {
    __threadfence();
    unsigned g = __hip_atomic_load(&bar[1], __ATOMIC_RELAXED, __HIP_MEMORY_SCOPE_AGENT);
    unsigned v = __hip_atomic_fetch_add(&bar[0], 1u, __ATOMIC_ACQ_REL, __HIP_MEMORY_SCOPE_AGENT);
    if (v == nblocks - 1u) {
      __hip_atomic_store(&bar[0], 0u, __ATOMIC_RELAXED, __HIP_MEMORY_SCOPE_AGENT);
      __hip_atomic_fetch_add(&bar[1], 1u, __ATOMIC_RELEASE, __HIP_MEMORY_SCOPE_AGENT);
    } else {
      while (__hip_atomic_load(&bar[1], __ATOMIC_ACQUIRE, __HIP_MEMORY_SCOPE_AGENT) == g) {
        __builtin_amdgcn_s_sleep(2);
      }
    }
    __threadfence();
  }
  __syncthreads();
}

// ------------------------------ WMMA GEMM ---------------------------------
// C[M,N] = A[M,K](f16) * W[N,K](f16)^T + bias[N].
// Grid: blockIdx.y -> 32-row A panel (staged through LDS in KCH chunks,
// shared by all waves of the block); each wave owns 64 output columns
// (8 v_wmma_f32_16x16x32_f16 per K-step, 8 independent accumulator chains).
// REMAP: out row m (= t*32+b time-major) -> (m%32)*SEQT + m/32 (batch-major).

template <bool F32OUT, bool REMAP>
__global__ __launch_bounds__(256) void k_gemm_wmma(
    const _Float16* __restrict__ A, const _Float16* __restrict__ W,
    const float* __restrict__ bias, void* __restrict__ out,
    int M, int N, int K) {
  __shared__ _Float16 as[32 * APAD];   // 16,896 B staged A panel chunk

  const int lane   = threadIdx.x & 31;
  const int wave   = threadIdx.x >> 5;
  const int nwaves = blockDim.x >> 5;
  const int mbase  = blockIdx.y << 5;
  const int nbase  = (blockIdx.x * nwaves + wave) << 6;
  const int krow   = lane >> 4;   // 0 or 1
  const int l16    = lane & 15;
  const bool active = (nbase < N);

  v8f acc[4][2];
  const _Float16* wp[4];
  if (active) {
#pragma unroll
    for (int nt = 0; nt < 4; ++nt) {
      int col = nbase + nt * 16 + l16;
      float bv = bias ? bias[col] : 0.0f;
#pragma unroll
      for (int r = 0; r < 8; ++r) { acc[nt][0][r] = bv; acc[nt][1][r] = bv; }
      wp[nt] = W + (size_t)col * K;
    }
  }

  for (int k0 = 0; k0 < K; k0 += KCH) {
    __syncthreads();   // previous chunk fully consumed
    // cooperative stage of A[mbase..+31][k0..k0+KCH) into padded LDS
    {
      const _Float16* ab = A + (size_t)mbase * K + k0;
      for (int c = threadIdx.x; c < 32 * (KCH / 8); c += blockDim.x) {
        int row = c / (KCH / 8);
        int ko  = (c % (KCH / 8)) * 8;
        *(v8h*)&as[row * APAD + ko] = *(const v8h*)(ab + (size_t)row * K + ko);
      }
    }
    __syncthreads();
    if (active) {
#pragma unroll
      for (int kk = 0; kk < KCH; kk += 32) {
        union { v16h v; v8h h[2]; } a0, a1;
        const _Float16* p0 = &as[l16 * APAD + kk + krow * 8];
        const _Float16* p1 = &as[(16 + l16) * APAD + kk + krow * 8];
        a0.h[0] = *(const v8h*)p0;  a0.h[1] = *(const v8h*)(p0 + 16);
        a1.h[0] = *(const v8h*)p1;  a1.h[1] = *(const v8h*)(p1 + 16);
#pragma unroll
        for (int nt = 0; nt < 4; ++nt) {
          union { v16h v; v8h h[2]; } bu;
          bu.h[0] = *(const v8h*)(wp[nt] + k0 + kk + krow * 8);
          bu.h[1] = *(const v8h*)(wp[nt] + k0 + kk + krow * 8 + 16);
          v16h bmat = bu.v;
          acc[nt][0] = __builtin_amdgcn_wmma_f32_16x16x32_f16(false, a0.v, false, bmat,
                                                              (short)0, acc[nt][0], false, false);
          acc[nt][1] = __builtin_amdgcn_wmma_f32_16x16x32_f16(false, a1.v, false, bmat,
                                                              (short)0, acc[nt][1], false, false);
        }
      }
    }
  }

  if (!active) return;
  int rbase = krow * 8;
#pragma unroll
  for (int nt = 0; nt < 4; ++nt) {
    int col = nbase + nt * 16 + l16;
#pragma unroll
    for (int r = 0; r < 8; ++r) {
      int m0 = mbase + rbase + r;
      int m1 = m0 + 16;
      int o0 = REMAP ? ((m0 & (BATCH - 1)) * SEQT + (m0 >> 5)) : m0;
      int o1 = REMAP ? ((m1 & (BATCH - 1)) * SEQT + (m1 >> 5)) : m1;
      if (F32OUT) {
        ((float*)out)[(size_t)o0 * N + col] = acc[nt][0][r];
        ((float*)out)[(size_t)o1 * N + col] = acc[nt][1][r];
      } else {
        ((_Float16*)out)[(size_t)o0 * N + col] = (_Float16)acc[nt][0][r];
        ((_Float16*)out)[(size_t)o1 * N + col] = (_Float16)acc[nt][1][r];
      }
    }
  }
}

// --------------------------- persistent recurrence ------------------------
// 32 blocks x 256 threads; block nb owns hidden units [nb*32, nb*32+32).
// Per wave: 16-column W_hh slice in 256 VGPRs for the whole sequence.
// Per step: one cooperative 64KB h(t-1) global->LDS stage, then the WMMA
// chain reads A tiles via ds_load from the padded LDS image.

__global__ __launch_bounds__(256) void k_lstm_recurrent(
    const _Float16* __restrict__ xW,   // [SEQT][BATCH][GATES] perm, bias folded
    const _Float16* __restrict__ Whh,  // [GATES][HDIM] perm rows, f16
    _Float16* __restrict__ hbuf,       // [2][BATCH][HDIM]
    _Float16* __restrict__ hs,         // [SEQT][BATCH][HDIM] layer output
    unsigned* __restrict__ bar,
    int nblocks) {
  __shared__ _Float16 hl[BATCH * HPAD];  // 66,048 B  staged h(t-1)
  __shared__ float    gl[BATCH * 128];   // 16,384 B  gate recombination

  const int tid  = threadIdx.x;
  const int lane = tid & 31;
  const int wave = tid >> 5;
  const int nb   = blockIdx.x;
  const int l16  = lane & 15;
  const int krow = lane >> 4;
  const int col  = nb * 128 + wave * 16 + l16;   // permuted gate col

  // Preload this wave's W_hh slice into registers: 32 x v16h = 256 VGPRs.
  const _Float16* wp = Whh + (size_t)col * HDIM;
  v16h wreg[32];
#pragma unroll
  for (int kk = 0; kk < 32; ++kk) {
    union { v16h v; v8h h[2]; } bu;
    bu.h[0] = *(const v8h*)(wp + kk * 32 + krow * 8);
    bu.h[1] = *(const v8h*)(wp + kk * 32 + krow * 8 + 16);
    wreg[kk] = bu.v;
  }

  // zero parity-0 h slice owned by this block
  for (int e = tid; e < BATCH * 32; e += blockDim.x) {
    int b = e >> 5, jl = e & 31;
    hbuf[(size_t)b * HDIM + nb * 32 + jl] = (_Float16)0.0f;
  }
  float creg[4] = {0.f, 0.f, 0.f, 0.f};
  grid_barrier(bar, (unsigned)nblocks);

  for (int t = 0; t < SEQT; ++t) {
    const _Float16* hin  = hbuf + (size_t)(t & 1) * BATCH * HDIM;
    _Float16*       hout = hbuf + (size_t)((t + 1) & 1) * BATCH * HDIM;

    // stage h(t-1): 4096 8-half chunks, 16 independent loads per thread
    {
      const v8h* src = (const v8h*)hin;
#pragma unroll
      for (int c0 = 0; c0 < 16; ++c0) {
        int c    = c0 * 256 + tid;
        int row  = c >> 7;            // 128 chunks per row
        int koff = (c & 127) * 8;
        *(v8h*)&hl[row * HPAD + koff] = src[c];
      }
    }

    // init accumulators with precomputed xW (time-major, bias folded in)
    v8f acc0, acc1;
    {
      const _Float16* xb = xW + (size_t)t * BATCH * GATES + col;
#pragma unroll
      for (int r = 0; r < 8; ++r) {
        int b0 = r + 8 * krow;
        acc0[r] = (float)xb[(size_t)b0 * GATES];
        acc1[r] = (float)xb[(size_t)(b0 + 16) * GATES];
      }
    }
    // prefetch next step's xW slice while this step computes
    if (t + 1 < SEQT) {
      const _Float16* xn = xW + (size_t)(t + 1) * BATCH * GATES + col;
#pragma unroll
      for (int r = 0; r < 8; ++r)
        __builtin_prefetch(xn + (size_t)(r + 8 * krow) * GATES, 0, 1);
    }
    __syncthreads();   // staged h visible

    // gates[:, col] += h(t-1) @ Whh[col,:]^T  (A from LDS, B from registers)
#pragma unroll
    for (int kk = 0; kk < 32; ++kk) {
      int k = kk * 32;
      union { v16h v; v8h h[2]; } a0, a1;
      const _Float16* p0 = &hl[l16 * HPAD + k + krow * 8];
      const _Float16* p1 = &hl[(16 + l16) * HPAD + k + krow * 8];
      a0.h[0] = *(const v8h*)p0;  a0.h[1] = *(const v8h*)(p0 + 16);
      a1.h[0] = *(const v8h*)p1;  a1.h[1] = *(const v8h*)(p1 + 16);
      acc0 = __builtin_amdgcn_wmma_f32_16x16x32_f16(false, a0.v, false, wreg[kk],
                                                    (short)0, acc0, false, false);
      acc1 = __builtin_amdgcn_wmma_f32_16x16x32_f16(false, a1.v, false, wreg[kk],
                                                    (short)0, acc1, false, false);
    }

    // scatter gate tile into LDS for cross-wave recombination
    int lc = wave * 16 + l16;
#pragma unroll
    for (int r = 0; r < 8; ++r) {
      int b0 = r + 8 * krow;
      gl[b0 * 128 + lc]        = acc0[r];
      gl[(b0 + 16) * 128 + lc] = acc1[r];
    }
    __syncthreads();

    // cell update: 32 batch x 32 hidden = 1024 items, 4 per thread
#pragma unroll
    for (int u = 0; u < 4; ++u) {
      int e  = tid * 4 + u;
      int b  = e >> 5;
      int jl = e & 31;
      float iv = sigmoidf_(gl[b * 128 + jl]);
      float fv = sigmoidf_(gl[b * 128 + 32 + jl]);
      float gv = tanhf(gl[b * 128 + 64 + jl]);
      float ov = sigmoidf_(gl[b * 128 + 96 + jl]);
      float c  = fv * creg[u] + iv * gv;
      creg[u]  = c;
      float h  = ov * tanhf(c);
      _Float16 h16 = (_Float16)h;
      int j = nb * 32 + jl;
      hout[(size_t)b * HDIM + j] = h16;
      hs[((size_t)t * BATCH + b) * HDIM + j] = h16;   // time-major
    }
    grid_barrier(bar, (unsigned)nblocks);
  }
}

// ------------------------------- launch -----------------------------------

extern "C" void kernel_launch(void* const* d_in, const int* in_sizes, int n_in,
                              void* d_out, int out_size, void* d_ws, size_t ws_size,
                              hipStream_t stream) {
  (void)in_sizes; (void)n_in; (void)out_size; (void)ws_size;
  const float* x    = (const float*)d_in[0];
  const float* Wih0 = (const float*)d_in[1];
  const float* Whh0 = (const float*)d_in[2];
  const float* bih0 = (const float*)d_in[3];
  const float* bhh0 = (const float*)d_in[4];
  const float* Wih1 = (const float*)d_in[5];
  const float* Whh1 = (const float*)d_in[6];
  const float* bih1 = (const float*)d_in[7];
  const float* bhh1 = (const float*)d_in[8];
  const float* fcw  = (const float*)d_in[9];
  const float* fcb  = (const float*)d_in[10];

  char* p = (char*)d_ws;
  auto alloc = [&](size_t bytes) -> char* {
    char* r = p;
    p += (bytes + 255) & ~(size_t)255;
    return r;
  };
  _Float16* xW    = (_Float16*)alloc((size_t)BATCH * SEQT * GATES * 2);  // 128 MB (reused)
  _Float16* hs0   = (_Float16*)alloc((size_t)BATCH * SEQT * HDIM * 2);   // 32 MB
  _Float16* hs1   = (_Float16*)alloc((size_t)BATCH * SEQT * HDIM * 2);   // 32 MB
  _Float16* x16   = (_Float16*)alloc((size_t)BATCH * SEQT * INDIM * 2);
  _Float16* wih0p = (_Float16*)alloc((size_t)GATES * INDIM * 2);
  _Float16* whh0p = (_Float16*)alloc((size_t)GATES * HDIM * 2);
  _Float16* wih1p = (_Float16*)alloc((size_t)GATES * HDIM * 2);
  _Float16* whh1p = (_Float16*)alloc((size_t)GATES * HDIM * 2);
  _Float16* fcw16 = (_Float16*)alloc((size_t)OUTDIM * HDIM * 2);
  float*    b0p   = (float*)alloc((size_t)GATES * 4);
  float*    b1p   = (float*)alloc((size_t)GATES * 4);
  _Float16* hbuf  = (_Float16*)alloc((size_t)2 * BATCH * HDIM * 2);
  unsigned* bar   = (unsigned*)alloc(256);

  k_init_barrier<<<1, 1, 0, stream>>>(bar);
  k_pack_x_tmajor<<<2048, 256, 0, stream>>>(x, x16);
  k_pack_f16<<<512, 256, 0, stream>>>(fcw, fcw16, (size_t)OUTDIM * HDIM);
  k_pack_w_perm<<<4096, 256, 0, stream>>>(Wih0, wih0p, INDIM);
  k_pack_w_perm<<<8192, 256, 0, stream>>>(Whh0, whh0p, HDIM);
  k_pack_w_perm<<<8192, 256, 0, stream>>>(Wih1, wih1p, HDIM);
  k_pack_w_perm<<<8192, 256, 0, stream>>>(Whh1, whh1p, HDIM);
  k_pack_bias<<<16, 256, 0, stream>>>(bih0, bhh0, b0p);
  k_pack_bias<<<16, 256, 0, stream>>>(bih1, bhh1, b1p);

  const int M = BATCH * SEQT;   // 16384 rows, time-major (t*32+b)
  const int nrec = HDIM / 32;   // 32 persistent blocks

  // layer 0: xW0 = x16 @ Wih0p^T + b0p  (f16 out, time-major identity)
  {
    dim3 g(GATES / 512, M / 32);   // 8 x 512
    k_gemm_wmma<false, false><<<g, 256, 0, stream>>>(x16, wih0p, b0p, xW, M, GATES, INDIM);
  }
  k_lstm_recurrent<<<nrec, 256, 0, stream>>>(xW, whh0p, hbuf, hs0, bar, nrec);

  // layer 1: xW1 = hs0 @ Wih1p^T + b1p  (reuse xW buffer)
  {
    dim3 g(GATES / 512, M / 32);
    k_gemm_wmma<false, false><<<g, 256, 0, stream>>>(hs0, wih1p, b1p, xW, M, GATES, HDIM);
  }
  k_lstm_recurrent<<<nrec, 256, 0, stream>>>(xW, whh1p, hbuf, hs1, bar, nrec);

  // final FC: out = hs1 @ fc_w^T + fc_b  (f32 out, remap rows to [b][t])
  {
    dim3 g(1, M / 32);             // 4 waves cover N=256
    k_gemm_wmma<true, true><<<g, 128, 0, stream>>>(hs1, fcw16, fcb, d_out, M, OUTDIM, HDIM);
  }
}
